// SelfAttention_2465311228213
// MI455X (gfx1250) — compile-verified, matrix operand order checked by reading, not run on hardware
//
#include <hip/hip_runtime.h>
#include <hip/hip_bf16.h>

#define SEQ 2048
#define DIM 512
#define NBATCH 8

typedef __attribute__((ext_vector_type(16))) __bf16        bf16x16;
typedef __attribute__((ext_vector_type(8)))  float         floatx8;
typedef __attribute__((ext_vector_type(4)))  float         floatx4;
typedef __attribute__((ext_vector_type(8)))  unsigned short ushortx8;
typedef __attribute__((ext_vector_type(16))) unsigned short ushortx16;

__device__ __forceinline__ unsigned short f32_to_bf16(float f) {
    unsigned u = __float_as_uint(f);
    u += 0x7FFFu + ((u >> 16) & 1u);   // round-to-nearest-even
    return (unsigned short)(u >> 16);
}

// Load one 16x32 bf16 WMMA A/B fragment from LDS (row-major, K-contiguous rows).
// Per ISA layout: lanes 0-15 hold row r, K {0..7,16..23}; lanes 16-31 row r, K {8..15,24..31}.
__device__ __forceinline__ bf16x16 ldsFrag(const unsigned short* base, int stride, int lane) {
    const int r  = lane & 15;
    const int ko = (lane >> 4) << 3;
    ushortx16 t;
    t.lo = *(const ushortx8*)(base + r * stride + ko);
    t.hi = *(const ushortx8*)(base + r * stride + ko + 16);
    return __builtin_bit_cast(bf16x16, t);
}

__device__ __forceinline__ floatx8 wmma_bf16(bf16x16 a, bf16x16 b, floatx8 c) {
    return __builtin_amdgcn_wmma_f32_16x16x32_bf16(false, a, false, b, (short)0, c, false, false);
}

// ---------------------------------------------------------------------------
// Kernel 1: QKV projection GEMM.  out = x(16384x512) * W(512x512) + bias.
// grid = (128, 4, 3): 128x128 output tile, z selects Q/K/V.
// Q,K stored bf16 row-major [B*S][D]; V stored transposed bf16 [B][D][S].
// ---------------------------------------------------------------------------
__global__ __launch_bounds__(256)
void qkv_proj_kernel(const float* __restrict__ x,
                     const float* __restrict__ Wq, const float* __restrict__ bq,
                     const float* __restrict__ Wk, const float* __restrict__ bk,
                     const float* __restrict__ Wv, const float* __restrict__ bv,
                     unsigned short* __restrict__ qb,
                     unsigned short* __restrict__ kb,
                     unsigned short* __restrict__ vt)
{
    constexpr int AS = 40;                     // padded bf16 row stride (128x32 tile)
    __shared__ __align__(16) unsigned short Al[128 * AS];
    __shared__ __align__(16) unsigned short Bl[128 * AS];

    const int tid  = threadIdx.x;
    const int lane = tid & 31;
    const int wave = tid >> 5;
    const int wm   = wave & 3;                 // 4 waves over M
    const int wn   = wave >> 2;                // 2 waves over N
    const int m0   = blockIdx.x * 128;
    const int n0   = blockIdx.y * 128;
    const int z    = blockIdx.z;

    const float* W    = (z == 0) ? Wq : (z == 1) ? Wk : Wv;
    const float* bias = (z == 0) ? bq : (z == 1) ? bk : bv;

    floatx8 acc[2][4];
    #pragma unroll
    for (int i = 0; i < 2; ++i)
        #pragma unroll
        for (int j = 0; j < 4; ++j) acc[i][j] = (floatx8)0.0f;

    const int lr = tid >> 1;                   // 0..127 tile row
    const int lc = (tid & 1) << 4;             // 0 or 16

    for (int ks = 0; ks < 16; ++ks) {
        // ---- stage A tile: x[m0+lr][ks*32+lc .. +15] f32 -> bf16 LDS
        {
            const float* xs = x + (size_t)(m0 + lr) * DIM + ks * 32 + lc;
            float fv[16];
            #pragma unroll
            for (int j = 0; j < 4; ++j) {
                floatx4 f = *(const floatx4*)(xs + j * 4);
                fv[j*4+0] = f.x; fv[j*4+1] = f.y; fv[j*4+2] = f.z; fv[j*4+3] = f.w;
            }
            ushortx8 h0, h1;
            #pragma unroll
            for (int i = 0; i < 8; ++i) { h0[i] = f32_to_bf16(fv[i]); h1[i] = f32_to_bf16(fv[8+i]); }
            *(ushortx8*)(Al + lr * AS + lc)     = h0;
            *(ushortx8*)(Al + lr * AS + lc + 8) = h1;
        }
        // ---- stage B tile: W^T[n0+lr][ks*32+lc .. +15] (strided reads of W)
        {
            const float* wsrc = W + (size_t)(ks * 32 + lc) * DIM + n0 + lr;
            ushortx8 g0, g1;
            #pragma unroll
            for (int i = 0; i < 8; ++i) {
                g0[i] = f32_to_bf16(wsrc[(size_t)i * DIM]);
                g1[i] = f32_to_bf16(wsrc[(size_t)(i + 8) * DIM]);
            }
            *(ushortx8*)(Bl + lr * AS + lc)     = g0;
            *(ushortx8*)(Bl + lr * AS + lc + 8) = g1;
        }
        __syncthreads();

        bf16x16 af0 = ldsFrag(Al + (wm * 32     ) * AS, AS, lane);
        bf16x16 af1 = ldsFrag(Al + (wm * 32 + 16) * AS, AS, lane);
        #pragma unroll
        for (int nf = 0; nf < 4; ++nf) {
            bf16x16 bf = ldsFrag(Bl + (wn * 64 + nf * 16) * AS, AS, lane);
            acc[0][nf] = wmma_bf16(af0, bf, acc[0][nf]);
            acc[1][nf] = wmma_bf16(af1, bf, acc[1][nf]);
        }
        __syncthreads();
    }

    // ---- epilogue: + bias, convert bf16, store
    const int lh = (lane >> 4) & 1;
    const int ln = lane & 15;
    #pragma unroll
    for (int mf = 0; mf < 2; ++mf)
        #pragma unroll
        for (int nf = 0; nf < 4; ++nf) {
            const int col = n0 + wn * 64 + nf * 16 + ln;
            const float bval = bias[col];
            #pragma unroll
            for (int v = 0; v < 8; ++v) {
                const int m = m0 + wm * 32 + mf * 16 + v + 8 * lh;
                const unsigned short h = f32_to_bf16(acc[mf][nf][v] + bval);
                if (z == 0)       qb[(size_t)m * DIM + col] = h;
                else if (z == 1)  kb[(size_t)m * DIM + col] = h;
                else {
                    const int bi = m >> 11;          // m / 2048
                    const int s  = m & 2047;
                    vt[((size_t)bi * DIM + col) * SEQ + s] = h;   // V transposed
                }
            }
        }
}

// ---------------------------------------------------------------------------
// Kernel 2: flash attention.  Block = 64 queries x full D=512, 8 waves.
// grid = (32, 8) = (S/64, B).  Online softmax over 32 key tiles of 64.
// ---------------------------------------------------------------------------
__global__ __launch_bounds__(256)
void attn_kernel(const unsigned short* __restrict__ qb,
                 const unsigned short* __restrict__ kb,
                 const unsigned short* __restrict__ vt,
                 float* __restrict__ out)
{
    constexpr int QS = 136;   // bf16 stride for 64x128 Q/K chunk (128+8 pad)
    constexpr int SS = 68;    // f32 stride for 64x64 scores (64+4 pad)
    constexpr int PS = 72;    // bf16 stride for 64x64 probs
    constexpr int VS = 72;    // bf16 stride for 128x64 V chunk

    // Aliased LDS regions: [0,17408)=Qs|S  [17408,34816)=Ks|P  [34816,53248)=Vs  +scale
    __shared__ __align__(16) char smem[17408 * 2 + 18432 + 256];
    unsigned short* Qs   = (unsigned short*)smem;
    float*          Sl   = (float*)smem;
    unsigned short* Ks   = (unsigned short*)(smem + 17408);
    unsigned short* Pl   = (unsigned short*)(smem + 17408);
    unsigned short* Vs   = (unsigned short*)(smem + 34816);
    float*          scLds = (float*)(smem + 34816 + 18432);

    const int tid  = threadIdx.x;
    const int lane = tid & 31;
    const int wave = tid >> 5;
    const int wm   = wave & 3;        // 4 waves over 64 queries (16 each)
    const int wn   = wave >> 2;       // 2 waves over keys (QK) / d-cols (PV)
    const int lh   = (lane >> 4) & 1;
    const int ln   = lane & 15;

    const int q0 = blockIdx.x * 64;
    const int b  = blockIdx.y;

    const unsigned short* Qbase = qb + (size_t)(b * SEQ + q0) * DIM;
    const unsigned short* Kbase = kb + (size_t)b * SEQ * DIM;
    const unsigned short* Vbase = vt + (size_t)b * DIM * SEQ;

    floatx8 acc[4][4];                 // [d-chunk][n-frag] -> wave's 16q x 256d of O
    #pragma unroll
    for (int i = 0; i < 4; ++i)
        #pragma unroll
        for (int j = 0; j < 4; ++j) acc[i][j] = (floatx8)0.0f;

    float m_r = -3.0e38f, l_r = 0.0f;  // per-row softmax state (valid in tid<64)

    const int qlr = tid >> 2;          // Q/K loader: row 0..63
    const int qlc = (tid & 3) << 5;    //             col group 0/32/64/96
    const int vlr = tid >> 1;          // V loader:   row 0..127
    const int vlc = (tid & 1) << 5;    //             col group 0/32

    for (int s0 = 0; s0 < SEQ; s0 += 64) {
        // ================= S = Q * K^T (64x64), K-loop over D in 4 chunks of 128
        floatx8 sAcc[2] = { (floatx8)0.0f, (floatx8)0.0f };
        for (int dc = 0; dc < 4; ++dc) {
            __syncthreads();           // protect Qs/Ks reuse
            #pragma unroll
            for (int j = 0; j < 4; ++j) {
                *(ushortx8*)(Qs + qlr * QS + qlc + j * 8) =
                    *(const ushortx8*)(Qbase + (size_t)qlr * DIM + dc * 128 + qlc + j * 8);
                *(ushortx8*)(Ks + qlr * QS + qlc + j * 8) =
                    *(const ushortx8*)(Kbase + (size_t)(s0 + qlr) * DIM + dc * 128 + qlc + j * 8);
            }
            __syncthreads();
            #pragma unroll
            for (int kc = 0; kc < 4; ++kc) {
                bf16x16 aq = ldsFrag(Qs + (wm * 16) * QS + kc * 32, QS, lane);
                #pragma unroll
                for (int nf = 0; nf < 2; ++nf) {
                    bf16x16 bkf = ldsFrag(Ks + (wn * 32 + nf * 16) * QS + kc * 32, QS, lane);
                    sAcc[nf] = wmma_bf16(aq, bkf, sAcc[nf]);
                }
            }
        }
        __syncthreads();               // Qs dead -> reuse as S
        #pragma unroll
        for (int nf = 0; nf < 2; ++nf)
            #pragma unroll
            for (int v = 0; v < 8; ++v)
                Sl[(wm * 16 + v + 8 * lh) * SS + wn * 32 + nf * 16 + ln] = sAcc[nf][v];
        __syncthreads();

        // ================= online softmax: thread r owns row r (tid<64)
        if (tid < 64) {
            const float* srow = Sl + tid * SS;
            float mx = -3.0e38f;
            for (int i = 0; i < 64; ++i) mx = fmaxf(mx, srow[i]);
            const float mn   = fmaxf(m_r, mx);
            const float corr = __expf(m_r - mn);
            float sum = 0.0f;
            unsigned short* prow = Pl + tid * PS;
            for (int i = 0; i < 64; ++i) {
                const float e = __expf(srow[i] - mn);
                sum += e;
                prow[i] = f32_to_bf16(e);
            }
            l_r = l_r * corr + sum;
            m_r = mn;
            scLds[tid] = corr;
        }
        __syncthreads();

        // ================= rescale O accumulators by exp(m_old - m_new)
        float sc[8];
        #pragma unroll
        for (int v = 0; v < 8; ++v) sc[v] = scLds[wm * 16 + v + 8 * lh];
        #pragma unroll
        for (int dc = 0; dc < 4; ++dc)
            #pragma unroll
            for (int nf = 0; nf < 4; ++nf)
                #pragma unroll
                for (int v = 0; v < 8; ++v) acc[dc][nf][v] *= sc[v];

        // ================= O += P * V, d in 4 chunks of 128
        bf16x16 ap0 = ldsFrag(Pl + (wm * 16) * PS,      PS, lane);
        bf16x16 ap1 = ldsFrag(Pl + (wm * 16) * PS + 32, PS, lane);
        for (int dc = 0; dc < 4; ++dc) {
            __syncthreads();           // protect Vs reuse
            #pragma unroll
            for (int j = 0; j < 4; ++j)
                *(ushortx8*)(Vs + vlr * VS + vlc + j * 8) =
                    *(const ushortx8*)(Vbase + (size_t)(dc * 128 + vlr) * SEQ + s0 + vlc + j * 8);
            __syncthreads();
            #pragma unroll
            for (int nf = 0; nf < 4; ++nf) {
                bf16x16 bv0 = ldsFrag(Vs + (wn * 64 + nf * 16) * VS,      VS, lane);
                bf16x16 bv1 = ldsFrag(Vs + (wn * 64 + nf * 16) * VS + 32, VS, lane);
                acc[dc][nf] = wmma_bf16(ap0, bv0, acc[dc][nf]);
                acc[dc][nf] = wmma_bf16(ap1, bv1, acc[dc][nf]);
            }
        }
        __syncthreads();
    }

    // ================= finalize: O /= l, store f32
    if (tid < 64) scLds[tid] = 1.0f / l_r;
    __syncthreads();
    float sci[8];
    #pragma unroll
    for (int v = 0; v < 8; ++v) sci[v] = scLds[wm * 16 + v + 8 * lh];

    float* ob = out + ((size_t)b * SEQ + q0) * DIM;
    #pragma unroll
    for (int dc = 0; dc < 4; ++dc)
        #pragma unroll
        for (int nf = 0; nf < 4; ++nf) {
            const int col = dc * 128 + wn * 64 + nf * 16 + ln;
            #pragma unroll
            for (int v = 0; v < 8; ++v) {
                const int row = wm * 16 + v + 8 * lh;
                ob[(size_t)row * DIM + col] = acc[dc][nf][v] * sci[v];
            }
        }
}

// ---------------------------------------------------------------------------
extern "C" void kernel_launch(void* const* d_in, const int* in_sizes, int n_in,
                              void* d_out, int out_size, void* d_ws, size_t ws_size,
                              hipStream_t stream) {
    const float* x  = (const float*)d_in[0];
    const float* Wq = (const float*)d_in[1];
    const float* bq = (const float*)d_in[2];
    const float* Wk = (const float*)d_in[3];
    const float* bk = (const float*)d_in[4];
    const float* Wv = (const float*)d_in[5];
    const float* bv = (const float*)d_in[6];

    const size_t elems = (size_t)NBATCH * SEQ * DIM;   // 16384*512
    unsigned short* qb = (unsigned short*)d_ws;
    unsigned short* kb = qb + elems;
    unsigned short* vt = kb + elems;

    qkv_proj_kernel<<<dim3(128, 4, 3), 256, 0, stream>>>(x, Wq, bq, Wk, bk, Wv, bv, qb, kb, vt);
    attn_kernel<<<dim3(SEQ / 64, NBATCH), 256, 0, stream>>>(qb, kb, vt, (float*)d_out);
}
